// PointMamba_60155311948114
// MI455X (gfx1250) — matrix operand order repaired and checked by reading, not compiled, not run on hardware
//
#include <hip/hip_runtime.h>
#include <hip/hip_bf16.h>
#include <math.h>

// ---- static model config ----
#define BD      4
#define NPTS    2048
#define GC      128
#define MNB     32
#define DMODEL  384
#define DINNER  768
#define NSTATE  16
#define DTRANK  24
#define LSEQ    384      // 3*G
#define NCLST   16
#define EPSLN   1e-5f

typedef __attribute__((ext_vector_type(16))) _Float16 v16h;
typedef __attribute__((ext_vector_type(8)))  float    v8f;
typedef unsigned int u32x4 __attribute__((ext_vector_type(4)));
typedef int          i32x4 __attribute__((ext_vector_type(4)));
typedef int          i32x8 __attribute__((ext_vector_type(8)));

#if defined(__HIP_DEVICE_COMPILE__) && __has_builtin(__builtin_amdgcn_tensor_load_to_lds) && __has_builtin(__builtin_amdgcn_s_wait_tensorcnt)
#define USE_TDM 1
#else
#define USE_TDM 0
#endif

// ============================================================================
// Generic fused WMMA GEMM:  C[M,N] = epi(A[M,K](lda) @ W[K,N](ldw))
// block = 128 threads (4 waves), tile 64x64, K stepped by 32 through LDS.
// TDM path: double-buffered tensor_load_to_lds so the DMA of tile i+1
// overlaps convert+WMMA of tile i (TENSORcnt completes in order -> wait<=2).
// ============================================================================
struct Epi {
  const float* bias;    // per-col bias (nullptr = none)
  const float* bng;     // eval-BN gamma (nullptr = no BN)
  const float* bnb;     // eval-BN beta
  const float* extra;   // per-batch extra bias [nbatch][N] (nullptr = none)
  int extra_rows;       // rows per batch for extra indexing (>=1)
  int act;              // 0 none, 1 relu, 2 gelu(tanh), 3 softplus, 4 leaky0.2
  int residual;         // 1 -> C += old C (residual add)
};

__device__ __forceinline__ float apply_act(float v, int act) {
  switch (act) {
    case 1: return v > 0.f ? v : 0.f;
    case 2: { float c = 0.7978845608028654f * (v + 0.044715f * v * v * v);
              return 0.5f * v * (1.f + tanhf(c)); }
    case 3: return (v > 20.f) ? v : log1pf(__expf(v));
    case 4: return v > 0.f ? v : 0.2f * v;
    default: return v;
  }
}

#if USE_TDM
// Program one TDM D# and DMA a rows x cols f32 tile (row stride = strideElems)
// from global into LDS at byte offset ldsOff. extRows/extCols are the remaining
// tensor extent from the tile origin: TDM zero-fills reads past them (handles
// all M/N/K tail cases for free).
__device__ __forceinline__ void tdm_load_2d(const float* gsrc, unsigned int ldsOff,
                                            int rows, int cols,
                                            int extRows, int extCols,
                                            int strideElems) {
  unsigned long long ga = (unsigned long long)(uintptr_t)gsrc;
  u32x4 g0;
  g0[0] = 1u;                                   // count=1, user mode
  g0[1] = ldsOff;                               // lds_addr (bytes)
  g0[2] = (unsigned int)ga;                     // global_addr[31:0]
  g0[3] = (unsigned int)((ga >> 32) & 0x1FFFFFFu) | (2u << 30);  // [56:32] | type=2
  unsigned int td0 = (unsigned int)extCols;     // tensor_dim0 (contiguous dim)
  unsigned int td1 = (unsigned int)extRows;     // tensor_dim1
  unsigned long long st0 = (unsigned long long)(unsigned int)strideElems;
  i32x8 g1;
  g1[0] = (int)(2u << 16);                      // data_size = 4 bytes
  g1[1] = (int)((td0 & 0xFFFFu) << 16);         // tensor_dim0[15:0] -> bits 63:48
  g1[2] = (int)((td0 >> 16) | ((td1 & 0xFFFFu) << 16));
  g1[3] = (int)((td1 >> 16) | ((unsigned int)cols << 16));   // tile_dim0
  g1[4] = (int)rows;                            // tile_dim1 (tile_dim2 = 0)
  g1[5] = (int)(unsigned int)st0;               // tensor_dim0_stride[31:0]
  g1[6] = (int)(unsigned int)(st0 >> 32);       // stride[47:32]; dim1_stride=0
  g1[7] = 0;
  i32x4 z4 = {0, 0, 0, 0};
  i32x8 z8 = {0, 0, 0, 0, 0, 0, 0, 0};
  // clang-23 / amdgpu-toolchain 6-arg form:
  // (g0[4], g1[8], g2[4], g3[4], extra[8], cpol)
  __builtin_amdgcn_tensor_load_to_lds(g0, g1, z4, z4, z8, 0);
}
#endif

// branch-free clamped tile loads into a 16-register buffer
__device__ __forceinline__ void loadA16(const float* __restrict__ A, int lda,
                                        int tm, int kb, int M, int K, int tid,
                                        float* __restrict__ v) {
  #pragma unroll
  for (int i = 0; i < 16; ++i) {
    int idx = tid + 128 * i;
    int k = idx & 31, r = idx >> 5;
    int gr = tm + r, gk = kb + k;
    int cr = gr < M ? gr : M - 1;
    int ck = gk < K ? gk : K - 1;
    float x = A[(size_t)cr * lda + ck];
    v[i] = (gr < M && gk < K) ? x : 0.f;
  }
}

__device__ __forceinline__ void loadB16(const float* __restrict__ W, int ldw,
                                        int tn, int kb, int N, int K, int tid,
                                        float* __restrict__ v) {
  #pragma unroll
  for (int i = 0; i < 16; ++i) {
    int idx = tid + 128 * i;
    int n = idx & 63, k = idx >> 6;
    int gn = tn + n, gk = kb + k;
    int cn = gn < N ? gn : N - 1;
    int ck = gk < K ? gk : K - 1;
    float x = W[(size_t)ck * ldw + cn];
    v[i] = (gn < N && gk < K) ? x : 0.f;
  }
}

__launch_bounds__(128)
__global__ void wmma_gemm(const float* __restrict__ A, int lda,
                          const float* __restrict__ W, int ldw,
                          float* __restrict__ C, int ldc,
                          int M, int N, int K, Epi e) {
  __shared__ _Float16 As[64][40];   // [row][k]
  __shared__ _Float16 Bs[64][40];   // [n][k]  (W staged transposed)
#if USE_TDM
  __shared__ float rawA[2][64][32]; // ping-pong TDM destinations
  __shared__ float rawB[2][32][64];
#endif
  const int tn   = blockIdx.x * 64;
  const int tm   = blockIdx.y * 64;
  const int tid  = threadIdx.x;
  const int lane = tid & 31;
  const int wave = tid >> 5;

  v8f acc[4] = {};

#if USE_TDM
  // prime the pipeline: tile 0 into buffer 0
  if (tid < 32) {
    tdm_load_2d(A + (size_t)tm * lda,
                (unsigned int)(uintptr_t)&rawA[0][0][0],
                64, 32, M - tm, K, lda);
    tdm_load_2d(W + tn,
                (unsigned int)(uintptr_t)&rawB[0][0][0],
                32, 64, K, N - tn, ldw);
  }
  int ping = 0;
  for (int kb = 0; kb < K; kb += 32, ping ^= 1) {
    const bool hasNext = (kb + 32 < K);
    if (tid < 32) {
      if (hasNext) {
        // issue tile i+1 into the other buffer, then wait only for tile i
        tdm_load_2d(A + (size_t)tm * lda + (kb + 32),
                    (unsigned int)(uintptr_t)&rawA[ping ^ 1][0][0],
                    64, 32, M - tm, K - (kb + 32), lda);
        tdm_load_2d(W + (size_t)(kb + 32) * ldw + tn,
                    (unsigned int)(uintptr_t)&rawB[ping ^ 1][0][0],
                    32, 64, K - (kb + 32), N - tn, ldw);
        __builtin_amdgcn_s_wait_tensorcnt(2);
      } else {
        __builtin_amdgcn_s_wait_tensorcnt(0);
      }
    }
    __syncthreads();
    // ---- convert raw f32 tiles -> f16 compute tiles; adjacent-K pairs so
    // the LDS traffic merges into b64 loads / b32 stores ----
    #pragma unroll
    for (int i = 0; i < 8; ++i) {
      int idx = tid + 128 * i;           // 0..1023 pairs
      int kp = idx & 15, r = idx >> 4;
      As[r][2 * kp]     = (_Float16)rawA[ping][r][2 * kp];
      As[r][2 * kp + 1] = (_Float16)rawA[ping][r][2 * kp + 1];
    }
    #pragma unroll
    for (int i = 0; i < 8; ++i) {
      int idx = tid + 128 * i;
      int n = idx & 63, kp = idx >> 6;
      Bs[n][2 * kp]     = (_Float16)rawB[ping][2 * kp][n];
      Bs[n][2 * kp + 1] = (_Float16)rawB[ping][2 * kp + 1][n];
    }
    __syncthreads();
#else
  float av[16], bv[16];
  loadA16(A, lda, tm, 0, M, K, tid, av);
  loadB16(W, ldw, tn, 0, N, K, tid, bv);
  for (int kb = 0; kb < K; kb += 32) {
    // ---- store staged registers into f16 LDS tiles ----
    #pragma unroll
    for (int i = 0; i < 16; ++i) {
      int idx = tid + 128 * i;
      int k = idx & 31, r = idx >> 5;
      As[r][k] = (_Float16)av[i];
    }
    #pragma unroll
    for (int i = 0; i < 16; ++i) {
      int idx = tid + 128 * i;
      int n = idx & 63, k = idx >> 6;
      Bs[n][k] = (_Float16)bv[i];
    }
    __syncthreads();
    // ---- pipeline: issue next tile's global loads behind the WMMA work ----
    if (kb + 32 < K) {
      __builtin_prefetch(&A[(size_t)(tm + (tid >> 1)) * lda + kb + 32], 0, 1);
      loadA16(A, lda, tm, kb + 32, M, K, tid, av);
      loadB16(W, ldw, tn, kb + 32, N, K, tid, bv);
    }
#endif
    // ---- A fragment: lane row = lane&15; K = 16*(lane>=16) + e ----
    const int arow = wave * 16 + (lane & 15);
    const int akb  = (lane < 16) ? 0 : 16;
    const _Float16* pa = &As[arow][akb];
    v16h af;
    #pragma unroll
    for (int t2 = 0; t2 < 8; ++t2) { af[t2] = pa[t2]; af[8 + t2] = pa[8 + t2]; }

    // ---- B fragments: lane col = lane&15; K = 16*(lane>=16) + e ----
    const int bn0 = lane & 15;
    const int bkb = (lane < 16) ? 0 : 16;
    #pragma unroll
    for (int j = 0; j < 4; ++j) {
      const _Float16* pb = &Bs[j * 16 + bn0][bkb];
      v16h bf;
      #pragma unroll
      for (int t2 = 0; t2 < 16; ++t2) bf[t2] = pb[t2];
      acc[j] = __builtin_amdgcn_wmma_f32_16x16x32_f16(
          false, af, false, bf, (short)0, acc[j], false, false);
    }
    __syncthreads();
  }

  // ---- epilogue: VGPR r -> M = r + 8*(lane>=16); col = lane&15 ----
  const int bn0   = lane & 15;
  const int mrow0 = tm + wave * 16 + ((lane >= 16) ? 8 : 0);
  #pragma unroll
  for (int j = 0; j < 4; ++j) {
    int col = tn + j * 16 + bn0;
    if (col >= N) continue;
    float bia = e.bias ? e.bias[col] : 0.f;
    float sc = 1.f, sb = 0.f;
    if (e.bng) { sc = e.bng[col] * rsqrtf(1.f + EPSLN); sb = e.bnb[col]; }
    #pragma unroll
    for (int r = 0; r < 8; ++r) {
      int row = mrow0 + r;
      if (row >= M) continue;
      float v = acc[j][r] + bia;
      if (e.extra) v += e.extra[(size_t)(row / e.extra_rows) * N + col];
      if (e.bng)  v = v * sc + sb;
      v = apply_act(v, e.act);
      if (e.residual) v += C[(size_t)row * ldc + col];
      C[(size_t)row * ldc + col] = v;
    }
  }
}

// ============================================================================
// Support kernels
// ============================================================================
__global__ void k_transpose_pts(const float* __restrict__ pts, float* __restrict__ pc) {
  int i = blockIdx.x * blockDim.x + threadIdx.x;
  if (i >= BD * NPTS * 3) return;
  int b = i / (NPTS * 3), rem = i % (NPTS * 3);
  int n = rem / 3, c = rem % 3;
  pc[i] = pts[(size_t)b * 3 * NPTS + (size_t)c * NPTS + n];
}

// Farthest point sampling: one block per batch; dists in LDS, deterministic
// argmax (first max wins, matching jnp.argmax).
__global__ void k_fps(const float* __restrict__ pc, int* __restrict__ fidx) {
  __shared__ float dists[NPTS];
  __shared__ float sval[256];
  __shared__ int   sidx[256];
  __shared__ float fpt[3];
  __shared__ int   sfar;
  int b = blockIdx.x, t = threadIdx.x;
  const float* p = pc + (size_t)b * NPTS * 3;
  for (int i = t; i < NPTS; i += 256) dists[i] = 1e10f;
  if (t == 0) sfar = 0;
  __syncthreads();
  for (int g = 0; g < GC; ++g) {
    int far = sfar;
    if (t == 0) {
      fidx[b * GC + g] = far;
      fpt[0] = p[far * 3]; fpt[1] = p[far * 3 + 1]; fpt[2] = p[far * 3 + 2];
    }
    __syncthreads();
    float bv = -1.f; int bi = 0x7fffffff;
    for (int i = t; i < NPTS; i += 256) {
      float dx = p[i * 3] - fpt[0], dy = p[i * 3 + 1] - fpt[1], dz = p[i * 3 + 2] - fpt[2];
      float m = fminf(dists[i], dx * dx + dy * dy + dz * dz);
      dists[i] = m;
      if (m > bv || (m == bv && i < bi)) { bv = m; bi = i; }
    }
    sval[t] = bv; sidx[t] = bi;
    __syncthreads();
    for (int off = 128; off; off >>= 1) {
      if (t < off) {
        float ov = sval[t + off]; int oi = sidx[t + off];
        if (ov > sval[t] || (ov == sval[t] && oi < sidx[t])) { sval[t] = ov; sidx[t] = oi; }
      }
      __syncthreads();
    }
    if (t == 0) sfar = sidx[0];
    __syncthreads();
  }
}

__global__ void k_center(const float* __restrict__ pc, const int* __restrict__ fidx,
                         float* __restrict__ center) {
  int i = blockIdx.x * blockDim.x + threadIdx.x;
  if (i >= BD * GC * 3) return;
  int b = i / (GC * 3), rem = i % (GC * 3);
  int g = rem / 3, c = rem % 3;
  center[i] = pc[((size_t)b * NPTS + fidx[b * GC + g]) * 3 + c];
}

// kNN top-M (smallest d2, ties -> lower index, emitted ascending): block per (b,g)
__global__ void k_knn(const float* __restrict__ pc, const float* __restrict__ center,
                      int* __restrict__ nidx) {
  __shared__ float d2[NPTS];
  __shared__ float sval[256];
  __shared__ int   sidx[256];
  int bg = blockIdx.x, b = bg / GC, t = threadIdx.x;
  const float* p = pc + (size_t)b * NPTS * 3;
  float cx = center[bg * 3], cy = center[bg * 3 + 1], cz = center[bg * 3 + 2];
  for (int i = t; i < NPTS; i += 256) {
    float dx = p[i * 3] - cx, dy = p[i * 3 + 1] - cy, dz = p[i * 3 + 2] - cz;
    d2[i] = dx * dx + dy * dy + dz * dz;
  }
  __syncthreads();
  for (int m = 0; m < MNB; ++m) {
    float bv = 3.4e38f; int bi = 0x7fffffff;
    for (int i = t; i < NPTS; i += 256) {
      float v = d2[i];
      if (v < bv || (v == bv && i < bi)) { bv = v; bi = i; }
    }
    sval[t] = bv; sidx[t] = bi;
    __syncthreads();
    for (int off = 128; off; off >>= 1) {
      if (t < off) {
        float ov = sval[t + off]; int oi = sidx[t + off];
        if (ov < sval[t] || (ov == sval[t] && oi < sidx[t])) { sval[t] = ov; sidx[t] = oi; }
      }
      __syncthreads();
    }
    if (t == 0) { nidx[(size_t)bg * MNB + m] = sidx[0]; d2[sidx[0]] = 3.4e38f; }
    __syncthreads();
  }
}

__global__ void k_group(const float* __restrict__ pc, const int* __restrict__ nidx,
                        const float* __restrict__ center, float* __restrict__ nb) {
  int i = blockIdx.x * blockDim.x + threadIdx.x;
  if (i >= BD * GC * MNB * 3) return;
  int c = i % 3, bm = i / 3, m = bm % MNB, bg = bm / MNB, b = bg / GC;
  nb[i] = pc[((size_t)b * NPTS + nidx[(size_t)bg * MNB + m]) * 3 + c] - center[bg * 3 + c];
}

__global__ void k_maxpool_m(const float* __restrict__ in, float* __restrict__ out, int C) {
  int i = blockIdx.x * blockDim.x + threadIdx.x;
  if (i >= BD * GC * C) return;
  int c = i % C, bg = i / C;
  const float* p = in + (size_t)bg * MNB * C + c;
  float v = -3.4e38f;
  for (int m = 0; m < MNB; ++m) v = fmaxf(v, p[(size_t)m * C]);
  out[i] = v;
}

__global__ void k_cat259(const float* __restrict__ gmax, const float* __restrict__ nb,
                         float* __restrict__ out) {
  int i = blockIdx.x * blockDim.x + threadIdx.x;
  if (i >= BD * GC * MNB * 259) return;
  int c = i % 259, bm = i / 259, m = bm % MNB, bg = bm / MNB;
  out[i] = (c < 256) ? gmax[(size_t)bg * 256 + c]
                     : nb[((size_t)bg * MNB + m) * 3 + (c - 256)];
}

// stable argsort of the k-th center coordinate (G=128): rank by count
__global__ void k_argsort_center(const float* __restrict__ center, int* __restrict__ inds) {
  __shared__ float v[GC];
  int bk = blockIdx.x, b = bk / 3, k = bk % 3, t = threadIdx.x;
  v[t] = center[((size_t)b * GC + t) * 3 + k];
  __syncthreads();
  float vi = v[t];
  int rank = 0;
  for (int j = 0; j < GC; ++j) {
    float vj = v[j];
    rank += (vj < vi) || (vj == vi && j < t);
  }
  inds[((size_t)b * 3 + k) * GC + rank] = t;
}

__global__ void k_seq(const float* __restrict__ tokens, const float* __restrict__ pos,
                      const int* __restrict__ inds, float* __restrict__ h) {
  int i = blockIdx.x * blockDim.x + threadIdx.x;
  if (i >= BD * LSEQ * DMODEL) return;
  int c = i % DMODEL, bs = i / DMODEL, s = bs % LSEQ, b = bs / LSEQ;
  int k = s / GC, g = s % GC;
  int src = inds[((size_t)b * 3 + k) * GC + g];
  size_t si = ((size_t)b * GC + src) * DMODEL + c;
  h[i] = tokens[si] + pos[si];
}

__global__ void k_seq3(const float* __restrict__ center, const int* __restrict__ inds,
                       float* __restrict__ c3) {
  int i = blockIdx.x * blockDim.x + threadIdx.x;
  if (i >= BD * LSEQ * 3) return;
  int c = i % 3, bs = i / 3, s = bs % LSEQ, b = bs / LSEQ;
  int k = s / GC, g = s % GC;
  int src = inds[((size_t)b * 3 + k) * GC + g];
  c3[i] = center[((size_t)b * GC + src) * 3 + c];
}

// LayerNorm over D=384, one wave per row; optional col-offset/ld for slicing
__global__ void k_layernorm(const float* __restrict__ in, const float* __restrict__ g,
                            const float* __restrict__ bp, float* __restrict__ out,
                            int rows, int outld, int outoff) {
  int w = (blockIdx.x * blockDim.x + threadIdx.x) >> 5;
  int lane = threadIdx.x & 31;
  if (w >= rows) return;
  const float* x = in + (size_t)w * DMODEL;
  float s = 0.f, s2 = 0.f;
  for (int c = lane; c < DMODEL; c += 32) { float v = x[c]; s += v; s2 += v * v; }
  for (int off = 16; off; off >>= 1) {
    s += __shfl_xor(s, off, 32);
    s2 += __shfl_xor(s2, off, 32);
  }
  float mean = s / DMODEL;
  float inv  = rsqrtf(s2 / DMODEL - mean * mean + EPSLN);
  float* o = out + (size_t)w * outld + outoff;
  for (int c = lane; c < DMODEL; c += 32)
    o[c] = (x[c] - mean) * inv * g[c] + bp[c];
}

// depthwise causal conv (k=4) + SiLU; x = first DINNER cols of xz
__global__ void k_conv_silu(const float* __restrict__ xz, const float* __restrict__ cw,
                            const float* __restrict__ cb, float* __restrict__ xc) {
  int i = blockIdx.x * blockDim.x + threadIdx.x;
  if (i >= BD * LSEQ * DINNER) return;
  int c = i % DINNER, bl = i / DINNER, l = bl % LSEQ, b = bl / LSEQ;
  float acc = cb[c];
  #pragma unroll
  for (int w = 0; w < 4; ++w) {
    int li = l - 3 + w;
    if (li >= 0) acc += xz[((size_t)b * LSEQ + li) * (2 * DINNER) + c] * cw[w * DINNER + c];
  }
  xc[i] = acc / (1.f + __expf(-acc));
}

// selective scan: thread per (b,d), NS=16 state in regs, sequential over L;
// fuses +D*x and *silu(z)
__global__ void k_scan(const float* __restrict__ dtb, const float* __restrict__ xc,
                       const float* __restrict__ dbc, const float* __restrict__ xz,
                       const float* __restrict__ Alog, const float* __restrict__ Dp,
                       float* __restrict__ y) {
  int i = blockIdx.x * blockDim.x + threadIdx.x;
  if (i >= BD * DINNER) return;
  int b = i / DINNER, d = i % DINNER;
  float Arow[NSTATE], hs[NSTATE];
  #pragma unroll
  for (int n = 0; n < NSTATE; ++n) { Arow[n] = -__expf(Alog[d * NSTATE + n]); hs[n] = 0.f; }
  float Dd = Dp[d];
  for (int l = 0; l < LSEQ; ++l) {
    size_t bl = (size_t)b * LSEQ + l;
    float dt = dtb[bl * DINNER + d];
    float xv = xc[bl * DINNER + d];
    float dtx = dt * xv;
    const float* bc = dbc + bl * 56;
    float acc = 0.f;
    #pragma unroll
    for (int n = 0; n < NSTATE; ++n) {
      hs[n] = __expf(dt * Arow[n]) * hs[n] + dtx * bc[DTRANK + n];
      acc += hs[n] * bc[DTRANK + NSTATE + n];
    }
    float z = xz[bl * (2 * DINNER) + DINNER + d];
    y[bl * DINNER + d] = (acc + Dd * xv) * (z / (1.f + __expf(-z)));
  }
}

// x_max / x_avg over sequence axis -> glob[:, 0:1152] and glob[:, 1152:2304]
__global__ void k_pool(const float* __restrict__ xf, float* __restrict__ glob) {
  int i = blockIdx.x * blockDim.x + threadIdx.x;
  if (i >= BD * 3 * DMODEL) return;
  int C = 3 * DMODEL;
  int c = i % C, b = i / C;
  const float* p = xf + (size_t)b * LSEQ * C + c;
  float mx = -3.4e38f, sm = 0.f;
  for (int l = 0; l < LSEQ; ++l) { float v = p[(size_t)l * C]; mx = fmaxf(mx, v); sm += v; }
  glob[(size_t)b * 2368 + c] = mx;
  glob[(size_t)b * 2368 + (3 * DMODEL) + c] = sm / LSEQ;
}

// label branch: leaky_relu(bn(cls @ w)) -> glob[:, 2304:2368]
__global__ void k_label(const float* __restrict__ cls, const float* __restrict__ w,
                        const float* __restrict__ g, const float* __restrict__ bb,
                        float* __restrict__ glob) {
  int i = blockIdx.x * blockDim.x + threadIdx.x;
  if (i >= BD * 64) return;
  int b = i / 64, c = i % 64;
  float acc = 0.f;
  for (int k = 0; k < NCLST; ++k) acc += cls[b * NCLST + k] * w[k * 64 + c];
  acc = acc * (g[c] * rsqrtf(1.f + EPSLN)) + bb[c];
  glob[(size_t)b * 2368 + 2304 + c] = acc > 0.f ? acc : 0.2f * acc;
}

// 3-NN inverse-distance interpolation: one wave per point
__global__ void k_interp(const float* __restrict__ pc, const float* __restrict__ c3,
                         const float* __restrict__ xf, float* __restrict__ out) {
  int wid = (blockIdx.x * blockDim.x + threadIdx.x) >> 5;
  int lane = threadIdx.x & 31;
  if (wid >= BD * NPTS) return;
  int b = wid / NPTS, n = wid % NPTS;
  const float* pp = pc + ((size_t)b * NPTS + n) * 3;
  float px = pp[0], py = pp[1], pz = pp[2];
  const float* cb = c3 + (size_t)b * LSEQ * 3;
  float dl[12];
  #pragma unroll
  for (int j = 0; j < 12; ++j) {
    int s = lane * 12 + j;
    float dx = px - cb[s * 3], dy = py - cb[s * 3 + 1], dz = pz - cb[s * 3 + 2];
    dl[j] = dx * dx + dy * dy + dz * dz;
  }
  int sel[3]; float wv[3];
  #pragma unroll
  for (int t = 0; t < 3; ++t) {
    float bv = 3.4e38f; int bi = 0x7fffffff;
    #pragma unroll
    for (int j = 0; j < 12; ++j) {
      int s = lane * 12 + j;
      bool taken = (t > 0 && s == sel[0]) || (t > 1 && s == sel[1]);
      float v = dl[j];
      if (!taken && (v < bv || (v == bv && s < bi))) { bv = v; bi = s; }
    }
    for (int off = 16; off; off >>= 1) {
      float ov = __shfl_xor(bv, off, 32);
      int   oi = __shfl_xor(bi, off, 32);
      if (ov < bv || (ov == bv && oi < bi)) { bv = ov; bi = oi; }
    }
    sel[t] = bi;
    wv[t] = 1.f / (fmaxf(bv, 0.f) + 1e-8f);
  }
  float ws = wv[0] + wv[1] + wv[2];
  wv[0] /= ws; wv[1] /= ws; wv[2] /= ws;
  const int C = 3 * DMODEL;
  const float* x0 = xf + ((size_t)b * LSEQ + sel[0]) * C;
  const float* x1 = xf + ((size_t)b * LSEQ + sel[1]) * C;
  const float* x2 = xf + ((size_t)b * LSEQ + sel[2]) * C;
  float* o = out + ((size_t)b * NPTS + n) * (C + 3);
  for (int c = lane; c < C; c += 32)
    o[c] = wv[0] * x0[c] + wv[1] * x1[c] + wv[2] * x2[c];
  if (lane < 3) o[C + lane] = pp[lane];
}

// ============================================================================
// Host orchestration
// ============================================================================
static inline Epi mkepi(const float* bias, const float* g, const float* b,
                        const float* extra, int er, int act, int res) {
  Epi e; e.bias = bias; e.bng = g; e.bnb = b; e.extra = extra;
  e.extra_rows = er; e.act = act; e.residual = res; return e;
}

static inline void gemm(hipStream_t s, const float* A, int lda, const float* W, int ldw,
                        float* C, int ldc, int M, int N, int K, Epi e) {
  dim3 g((N + 63) / 64, (M + 63) / 64);
  wmma_gemm<<<g, 128, 0, s>>>(A, lda, W, ldw, C, ldc, M, N, K, e);
}

#define EWG(n) dim3(((n) + 255) / 256)

extern "C" void kernel_launch(void* const* d_in, const int* in_sizes, int n_in,
                              void* d_out, int out_size, void* d_ws, size_t ws_size,
                              hipStream_t stream) {
  auto F = [&](int i) { return (const float*)d_in[i]; };
  // input indices (insertion order of setup_inputs)
  const float* pts = F(0);
  const float* cls = F(1);
  // enc: 2..13  pos: 14..17  layers: 18 + 11*l + j  post_norm: 106,107
  // label: 108..110  prop: 111..118  head: 119..128
  auto Lp = [&](int l, int j) { return F(18 + l * 11 + j); };

  float* ws = (float*)d_ws;
  size_t off = 0;
  auto alloc = [&](size_t n) { float* p = ws + off; off += n; return p; };

  float* pc      = alloc((size_t)BD * NPTS * 3);
  float* center  = alloc((size_t)BD * GC * 3);
  float* center3 = alloc((size_t)BD * LSEQ * 3);
  int*   fidx    = (int*)alloc((size_t)BD * GC);
  int*   nidx    = (int*)alloc((size_t)BD * GC * MNB);
  int*   inds    = (int*)alloc((size_t)BD * 3 * GC);
  float* glob    = alloc((size_t)BD * 2368);
  float* gw1     = alloc((size_t)BD * 512);
  float* tokens  = alloc((size_t)BD * GC * DMODEL);
  float* posh    = alloc((size_t)BD * GC * 128);
  float* posb    = alloc((size_t)BD * GC * DMODEL);
  float* hbuf    = alloc((size_t)BD * LSEQ * DMODEL);
  float* ubuf    = alloc((size_t)BD * LSEQ * DMODEL);
  float* xzbuf   = alloc((size_t)BD * LSEQ * 2 * DINNER);
  float* xcbuf   = alloc((size_t)BD * LSEQ * DINNER);
  float* dbcbuf  = alloc((size_t)BD * LSEQ * 56);
  float* dtbuf   = alloc((size_t)BD * LSEQ * DINNER);
  float* ybuf    = alloc((size_t)BD * LSEQ * DINNER);
  float* xf      = alloc((size_t)BD * LSEQ * 3 * DMODEL);
  // region R: encoder temporaries, later reused by propagation/head stage
  float* R    = ws + off;
  float* nb   = R;
  float* f1   = nb   + (size_t)BD * GC * MNB * 3;
  float* gmax = f1   + (size_t)BD * GC * MNB * 128;
  float* f2   = gmax + (size_t)BD * GC * 256;
  float* cat  = f2   + (size_t)BD * GC * MNB * 256;
  float* f3   = cat  + (size_t)BD * GC * MNB * 259;
  float* f4   = f3   + (size_t)BD * GC * MNB * 512;
  // reuse of R after tokens are built:
  float* itp = R;
  float* p1  = itp + (size_t)BD * NPTS * 1155;
  float* p2  = p1  + (size_t)BD * NPTS * 1536;
  float* h1  = p2  + (size_t)BD * NPTS * 1024;
  float* h2  = h1  + (size_t)BD * NPTS * 512;

  const int BGM = BD * GC * MNB;   // 16384
  const int BL  = BD * LSEQ;       // 1536
  const int BN_ = BD * NPTS;       // 8192
  Epi none = mkepi(nullptr, nullptr, nullptr, nullptr, 1, 0, 0);

  // ---- grouping ----
  k_transpose_pts<<<EWG(BD * NPTS * 3), 256, 0, stream>>>(pts, pc);
  k_fps<<<BD, 256, 0, stream>>>(pc, fidx);
  k_center<<<EWG(BD * GC * 3), 256, 0, stream>>>(pc, fidx, center);
  k_knn<<<BD * GC, 256, 0, stream>>>(pc, center, nidx);
  k_group<<<EWG(BGM * 3), 256, 0, stream>>>(pc, nidx, center, nb);

  // ---- encoder ----
  gemm(stream, nb, 3, F(2), 128, f1, 128, BGM, 128, 3,
       mkepi(F(3), F(4), F(5), nullptr, 1, 1, 0));
  gemm(stream, f1, 128, F(6), 256, f2, 256, BGM, 256, 128,
       mkepi(F(7), nullptr, nullptr, nullptr, 1, 0, 0));
  k_maxpool_m<<<EWG(BD * GC * 256), 256, 0, stream>>>(f2, gmax, 256);
  k_cat259<<<EWG(BGM * 259), 256, 0, stream>>>(gmax, nb, cat);
  gemm(stream, cat, 259, F(8), 512, f3, 512, BGM, 512, 259,
       mkepi(F(9), F(10), F(11), nullptr, 1, 1, 0));
  gemm(stream, f3, 512, F(12), DMODEL, f4, DMODEL, BGM, DMODEL, 512,
       mkepi(F(13), nullptr, nullptr, nullptr, 1, 0, 0));
  k_maxpool_m<<<EWG(BD * GC * DMODEL), 256, 0, stream>>>(f4, tokens, DMODEL);

  // ---- positional embedding ----
  gemm(stream, center, 3, F(14), 128, posh, 128, BD * GC, 128, 3,
       mkepi(F(15), nullptr, nullptr, nullptr, 1, 2, 0));
  gemm(stream, posh, 128, F(16), DMODEL, posb, DMODEL, BD * GC, DMODEL, 128,
       mkepi(F(17), nullptr, nullptr, nullptr, 1, 0, 0));

  // ---- xyz-sort, concat, h = tokens + pos ----
  k_argsort_center<<<BD * 3, GC, 0, stream>>>(center, inds);
  k_seq<<<EWG(BL * DMODEL), 256, 0, stream>>>(tokens, posb, inds, hbuf);
  k_seq3<<<EWG(BL * 3), 256, 0, stream>>>(center, inds, center3);

  // ---- Mamba backbone (layer 8 feeds nothing downstream -> skipped) ----
  int fi = 0;
  for (int l = 0; l < 7; ++l) {
    k_layernorm<<<EWG(BL * 32), 256, 0, stream>>>(hbuf, Lp(l, 0), Lp(l, 1),
                                                  ubuf, BL, DMODEL, 0);
    gemm(stream, ubuf, DMODEL, Lp(l, 2), 2 * DINNER, xzbuf, 2 * DINNER,
         BL, 2 * DINNER, DMODEL, none);
    k_conv_silu<<<EWG(BL * DINNER), 256, 0, stream>>>(xzbuf, Lp(l, 3), Lp(l, 4), xcbuf);
    gemm(stream, xcbuf, DINNER, Lp(l, 5), 56, dbcbuf, 56, BL, 56, DINNER, none);
    gemm(stream, dbcbuf, 56, Lp(l, 6), DINNER, dtbuf, DINNER, BL, DINNER, DTRANK,
         mkepi(Lp(l, 7), nullptr, nullptr, nullptr, 1, 3, 0));
    k_scan<<<EWG(BD * DINNER), 256, 0, stream>>>(dtbuf, xcbuf, dbcbuf, xzbuf,
                                                 Lp(l, 8), Lp(l, 9), ybuf);
    gemm(stream, ybuf, DINNER, Lp(l, 10), DMODEL, hbuf, DMODEL, BL, DMODEL, DINNER,
         mkepi(nullptr, nullptr, nullptr, nullptr, 1, 0, 1));   // residual
    if (l == 0 || l == 2 || l == 6) {
      k_layernorm<<<EWG(BL * 32), 256, 0, stream>>>(hbuf, F(106), F(107), xf,
                                                    BL, 3 * DMODEL, fi * DMODEL);
      ++fi;
    }
  }

  // ---- global feature ----
  k_pool<<<EWG(BD * 3 * DMODEL), 256, 0, stream>>>(xf, glob);
  k_label<<<1, 256, 0, stream>>>(cls, F(108), F(109), F(110), glob);
  // fold glob @ head.w1[1024:3392] into per-batch bias gw1
  gemm(stream, glob, 2368, F(119) + (size_t)1024 * 512, 512, gw1, 512,
       BD, 512, 2368, none);

  // ---- feature propagation + seg head ----
  k_interp<<<EWG(BN_ * 32), 256, 0, stream>>>(pc, center3, xf, itp);
  gemm(stream, itp, 1155, F(111), 1536, p1, 1536, BN_, 1536, 1155,
       mkepi(F(112), F(113), F(114), nullptr, 1, 1, 0));
  gemm(stream, p1, 1536, F(115), 1024, p2, 1024, BN_, 1024, 1536,
       mkepi(F(116), F(117), F(118), nullptr, 1, 1, 0));
  gemm(stream, p2, 1024, F(119), 512, h1, 512, BN_, 512, 1024,
       mkepi(F(120), F(121), F(122), gw1, NPTS, 1, 0));
  gemm(stream, h1, 512, F(123), 256, h2, 256, BN_, 256, 512,
       mkepi(F(124), F(125), F(126), nullptr, 1, 1, 0));
  gemm(stream, h2, 256, F(127), NCLST, (float*)d_out, NCLST, BN_, NCLST, 256,
       mkepi(F(128), nullptr, nullptr, nullptr, 1, 0, 0));
}
